// GINConv_3693671875302
// MI455X (gfx1250) — matrix operand order repaired and checked by reading, not compiled, not run on hardware
//
#include <hip/hip_runtime.h>

// ---------------------------------------------------------------------------
// GINConv (D=64): h = (1+eps)*x + scatter_sum(x[src] -> dst)
//                 out = relu(h@W1 + b1) @ W2 + b2
// Phases:
//   1) gin_init    : h = (1+eps)*x                     (memory-bound, float4)
//   2) gin_prepw   : W1,W2 f32 -> f16, pre-swizzled into WMMA B-fragment layout
//   3) gin_scatter : atomicAdd gather/scatter over edges (L2-resident)
//   4) gin_mlp     : fused 2-layer MLP, v_wmma_f32_16x16x32_f16
// ---------------------------------------------------------------------------

typedef __attribute__((ext_vector_type(16))) _Float16 v16h;
typedef __attribute__((ext_vector_type(8)))  float    v8f;
typedef __attribute__((ext_vector_type(4)))  float    v4f;

#define D64 64
#define LSTRIDE 72  // padded LDS row stride (halves) to dodge bank conflicts

// ---- phase 1: h = (1+eps)*x -----------------------------------------------
__global__ void gin_init(const float* __restrict__ x,
                         const float* __restrict__ eps,
                         float* __restrict__ h, int n4) {
    int i = blockIdx.x * blockDim.x + threadIdx.x;
    if (i >= n4) return;
    float s = 1.0f + eps[0];
    v4f v = ((const v4f*)x)[i];
    ((v4f*)h)[i] = v * s;
}

// ---- phase 2: pre-swizzle weights into WMMA B-fragment layout --------------
// Fragment (nb, ks): B element (k, n) with k = ks*32 + (lane>>4)*8 + s + (s>=8 ? 8 : 0),
//                    n = nb*16 + (lane&15)   (ISA 7.12.2 16-bit B layout)
// Stored: Wh[((nb*2+ks)*32 + lane)*16 + s]  -> contiguous 32B per lane.
__global__ void gin_prepw(const float* __restrict__ W1,
                          const float* __restrict__ W2,
                          _Float16* __restrict__ W1h,
                          _Float16* __restrict__ W2h) {
    int t = blockIdx.x * blockDim.x + threadIdx.x;   // 0..511
    const float* W  = (t < 256) ? W1 : W2;
    _Float16*    Wh = (t < 256) ? W1h : W2h;
    int r    = t & 255;
    int frag = r >> 5;            // nb*2 + ks
    int lane = r & 31;
    int ks   = frag & 1;
    int nb   = frag >> 1;
    int n    = nb * 16 + (lane & 15);
    int hh   = lane >> 4;
    _Float16* dst = Wh + ((size_t)(frag * 32 + lane)) * 16;
    #pragma unroll
    for (int s = 0; s < 16; ++s) {
        int k = ks * 32 + hh * 8 + s + (s >= 8 ? 8 : 0);
        dst[s] = (_Float16)W[k * D64 + n];
    }
}

// ---- phase 3: edge scatter-sum (one wave per edge, float2 per lane) --------
__global__ void gin_scatter(const float* __restrict__ x,
                            const long long* __restrict__ ei, // int64 [2,E]
                            float* __restrict__ h, long long E) {
    long long t = (long long)blockIdx.x * blockDim.x + threadIdx.x;
    long long e = t >> 5;
    if (e >= E) return;
    int d = (int)(t & 31) * 2;
    long long src = ei[e];
    long long dst = ei[E + e];
    const float* xs = x + src * D64 + d;
    float v0 = xs[0], v1 = xs[1];
    float* hd = h + dst * D64 + d;
    atomicAdd(hd,     v0);
    atomicAdd(hd + 1, v1);
}

// ---- A-fragment load from an f32 row buffer (ISA 7.12.2 16-bit A layout) ---
// lane = hh*16 + m ; slots 0..7 -> k = ks*32 + hh*8 + j
//                    slots 8..15 -> k = ks*32 + 16 + hh*8 + j
__device__ inline v16h loadA_f32(const float* __restrict__ row, int ks, int hh) {
    int base = ks * 32 + hh * 8;
    v4f p0 = *(const v4f*)(row + base);
    v4f p1 = *(const v4f*)(row + base + 4);
    v4f p2 = *(const v4f*)(row + base + 16);
    v4f p3 = *(const v4f*)(row + base + 20);
    v16h a;
    #pragma unroll
    for (int j = 0; j < 4; ++j) {
        a[j]      = (_Float16)p0[j];
        a[4 + j]  = (_Float16)p1[j];
        a[8 + j]  = (_Float16)p2[j];
        a[12 + j] = (_Float16)p3[j];
    }
    return a;
}

// ---- phase 4: fused MLP; one wave per 16-row tile, 4 waves per block -------
__global__ __launch_bounds__(128, 1) void gin_mlp(
        const float* __restrict__ h,
        const _Float16* __restrict__ W1h,
        const _Float16* __restrict__ W2h,
        const float* __restrict__ b1,
        const float* __restrict__ b2,
        float* __restrict__ out, int nrows) {
    __shared__ _Float16 smem[4][16 * LSTRIDE];
    const int lane = threadIdx.x & 31;
    const int warp = threadIdx.x >> 5;
    const int tile = blockIdx.x * 4 + warp;
    const int row0 = tile * 16;
    if (row0 >= nrows) return;              // wave-uniform: EXEC stays all-1s

    const int m16 = lane & 15;              // A row / B,C column within tile
    const int hh  = lane >> 4;
    _Float16* tl = smem[warp];

    // ---------------- GEMM1: acc = h_tile @ W1 + b1 ----------------
    v8f acc[4];
    #pragma unroll
    for (int nb = 0; nb < 4; ++nb) {
        float bv = b1[nb * 16 + m16];
        #pragma unroll
        for (int v = 0; v < 8; ++v) acc[nb][v] = bv;
    }
    const float* hrow = h + (size_t)(row0 + m16) * D64;
    #pragma unroll
    for (int ks = 0; ks < 2; ++ks) {
        v16h a = loadA_f32(hrow, ks, hh);
        #pragma unroll
        for (int nb = 0; nb < 4; ++nb) {
            v16h b = *(const v16h*)(W1h + (size_t)((nb * 2 + ks) * 32 + lane) * 16);
            acc[nb] = __builtin_amdgcn_wmma_f32_16x16x32_f16(
                false, a, false, b, (short)0, acc[nb], false, false);
        }
    }

    // ---------------- ReLU, C-layout -> LDS (row-major f16 tile) ----------
    #pragma unroll
    for (int nb = 0; nb < 4; ++nb) {
        #pragma unroll
        for (int v = 0; v < 8; ++v) {
            int m = v + hh * 8;
            tl[m * LSTRIDE + nb * 16 + m16] = (_Float16)fmaxf(acc[nb][v], 0.0f);
        }
    }
    // single wave: DS ops are in-order; compiler inserts s_wait_dscnt on reads

    // ---------------- GEMM2: acc = relu_tile @ W2 + b2 --------------------
    #pragma unroll
    for (int nb = 0; nb < 4; ++nb) {
        float bv = b2[nb * 16 + m16];
        #pragma unroll
        for (int v = 0; v < 8; ++v) acc[nb][v] = bv;
    }
    const _Float16* tr = tl + m16 * LSTRIDE;
    #pragma unroll
    for (int ks = 0; ks < 2; ++ks) {
        int base = ks * 32 + hh * 8;
        v16h a;
        #pragma unroll
        for (int j = 0; j < 8; ++j) {
            a[j]     = tr[base + j];
            a[8 + j] = tr[base + 16 + j];
        }
        #pragma unroll
        for (int nb = 0; nb < 4; ++nb) {
            v16h b = *(const v16h*)(W2h + (size_t)((nb * 2 + ks) * 32 + lane) * 16);
            acc[nb] = __builtin_amdgcn_wmma_f32_16x16x32_f16(
                false, a, false, b, (short)0, acc[nb], false, false);
        }
    }

    // ---------------- store C (f32): element (m,n) -> out[row0+m][nb*16+n] -
    #pragma unroll
    for (int nb = 0; nb < 4; ++nb) {
        #pragma unroll
        for (int v = 0; v < 8; ++v) {
            int m = v + hh * 8;
            out[(size_t)(row0 + m) * D64 + nb * 16 + m16] = acc[nb][v];
        }
    }
}

extern "C" void kernel_launch(void* const* d_in, const int* in_sizes, int n_in,
                              void* d_out, int out_size, void* d_ws, size_t ws_size,
                              hipStream_t stream) {
    const float*     x   = (const float*)d_in[0];
    const long long* ei  = (const long long*)d_in[1]; // int64 per reference
    // d_in[2] = edge_attr (unused by GINConv)
    const float*     eps = (const float*)d_in[3];
    const float*     W1  = (const float*)d_in[4];
    const float*     b1  = (const float*)d_in[5];
    const float*     W2  = (const float*)d_in[6];
    const float*     b2  = (const float*)d_in[7];
    float*           out = (float*)d_out;

    const int       N = in_sizes[0] / D64;       // 50000
    const long long E = (long long)in_sizes[1] / 2; // 800000

    // workspace layout: [ h : N*64 f32 ][ W1h : 4096 f16 ][ W2h : 4096 f16 ]
    float*    h   = (float*)d_ws;
    size_t    hb  = (size_t)N * D64 * sizeof(float);
    _Float16* W1h = (_Float16*)((char*)d_ws + hb);
    _Float16* W2h = W1h + 8 * 32 * 16;

    // 1) h = (1+eps)*x
    int n4 = N * D64 / 4;
    gin_init<<<(n4 + 255) / 256, 256, 0, stream>>>(x, eps, h, n4);

    // 2) weight pre-swizzle (f32 -> f16, WMMA B-fragment layout)
    gin_prepw<<<2, 256, 0, stream>>>(W1, W2, W1h, W2h);

    // 3) edge scatter-sum (one wave per edge)
    long long sthreads = E * 32;
    gin_scatter<<<(unsigned)((sthreads + 255) / 256), 256, 0, stream>>>(x, ei, h, E);

    // 4) fused WMMA MLP
    int ntiles = (N + 15) / 16;                  // 3125
    gin_mlp<<<(ntiles + 3) / 4, 128, 0, stream>>>(h, W1h, W2h, b1, b2, out, N);
}
